// RandLA_66022237274590
// MI455X (gfx1250) — matrix operand order repaired and checked by reading, not compiled, not run on hardware
//
#include <hip/hip_runtime.h>
#include <hip/hip_bf16.h>

typedef _Float16 v16h __attribute__((ext_vector_type(16)));
typedef float    v8f  __attribute__((ext_vector_type(8)));

#define BB   4
#define NN   65536
#define KK   16
#define NEGS 0.2f
#define EPSF 1e-5f

__device__ __forceinline__ float lrelu(float x) { return x > 0.f ? x : NEGS * x; }

// ---- cross-lane xor-butterfly within 16-lane rows: v_permlane16_b32 (VALU) ----
#if __has_builtin(__builtin_amdgcn_permlane16)
__device__ __forceinline__ float plx_(float v, unsigned s0, unsigned s1) {
    int u = __float_as_int(v);
    u = __builtin_amdgcn_permlane16(u, u, (int)s0, (int)s1, false, false);
    return __int_as_float(u);
}
__device__ __forceinline__ float plxor(float v, int m) {
    switch (m) {
        case 1:  return plx_(v, 0x67452301u, 0xEFCDAB89u);
        case 2:  return plx_(v, 0x54761032u, 0xDCFE98BAu);
        case 4:  return plx_(v, 0x32107654u, 0xBA98FEDCu);
        default: return plx_(v, 0xFEDCBA98u, 0x76543210u);
    }
}
#else
__device__ __forceinline__ float plxor(float v, int m) { return __shfl_xor(v, m, 32); }
#endif

__device__ __forceinline__ float rowsum16(float v) {
    v += plxor(v, 1); v += plxor(v, 2); v += plxor(v, 4); v += plxor(v, 8);
    return v;
}
__device__ __forceinline__ float rowmax16(float v) {
    v = fmaxf(v, plxor(v, 1)); v = fmaxf(v, plxor(v, 2));
    v = fmaxf(v, plxor(v, 4)); v = fmaxf(v, plxor(v, 8));
    return v;
}

struct P {
    const float* feature; const float* xyz; const int* nidx;
    const float *W_mlp1,*b_mlp1,*g_mlp1,*be_mlp1;
    const float *W_lfa1,*b_lfa1,*g_lfa1,*be_lfa1;
    const float *Wa1,*Wm1,*bm1,*gm1,*bem1;
    const float *W_lfa2,*b_lfa2,*g_lfa2,*be_lfa2;
    const float *Wa2,*Wm2,*bm2,*gm2,*bem2;
    const float *W_mlp2,*b_mlp2,*g_mlp2,*be_mlp2;
    const float *W_skip,*b_skip,*g_skip,*be_skip;
    float *fpc,*shc,*fagg,*fagg2,*fout;
    float *sraw,*sfin;
    float *out;
};

// ---------------- stage 1: f_pc_pre, shortcut_pre, GN1/GN2/GN7 stats ----------------
__global__ void k_stage1(P p) {
    __shared__ float sac[24];
    const int tid = threadIdx.x;
    if (tid < 24) sac[tid] = 0.f;
    __syncthreads();

    const int b = blockIdx.x >> 8;
    const int n = ((blockIdx.x & 255) << 8) + tid;

    float f[8];
#pragma unroll
    for (int c = 0; c < 8; ++c) f[c] = p.feature[(b * 8 + c) * NN + n];

    float s1[4] = {0,0,0,0}, q1[4] = {0,0,0,0};
#pragma unroll
    for (int o = 0; o < 8; ++o) {
        float v = p.b_mlp1[o];
#pragma unroll
        for (int c = 0; c < 8; ++c) v += p.W_mlp1[o * 8 + c] * f[c];
        p.fpc[(b * 8 + o) * NN + n] = v;
        s1[o >> 1] += v; q1[o >> 1] += v * v;
    }

    float s7[4] = {0,0,0,0}, q7[4] = {0,0,0,0};
#pragma unroll
    for (int o = 0; o < 32; ++o) {
        float v = p.b_skip[o];
#pragma unroll
        for (int c = 0; c < 8; ++c) v += p.W_skip[o * 8 + c] * f[c];
        p.shc[(b * 32 + o) * NN + n] = v;
        s7[o >> 3] += v; q7[o >> 3] += v * v;
    }

    const float px = p.xyz[(b * 3 + 0) * NN + n];
    const float py = p.xyz[(b * 3 + 1) * NN + n];
    const float pz = p.xyz[(b * 3 + 2) * NN + n];

    float s2[4] = {0,0,0,0}, q2[4] = {0,0,0,0};
    for (int k = 0; k < KK; ++k) {
        const int idx = p.nidx[(b * NN + n) * KK + k];
        const float nx = p.xyz[(b * 3 + 0) * NN + idx];
        const float ny = p.xyz[(b * 3 + 1) * NN + idx];
        const float nz = p.xyz[(b * 3 + 2) * NN + idx];
        const float rx = nx - px, ry = ny - py, rz = nz - pz;
        const float dist = sqrtf(rx * rx + ry * ry + rz * rz);
        const float frel[10] = {dist, rx, ry, rz, px, py, pz, nx, ny, nz};
#pragma unroll
        for (int c2 = 0; c2 < 8; ++c2) {
            float v = p.b_lfa1[c2];
#pragma unroll
            for (int j = 0; j < 10; ++j) v += p.W_lfa1[c2 * 10 + j] * frel[j];
            s2[c2 >> 1] += v; q2[c2 >> 1] += v * v;
        }
    }

#pragma unroll
    for (int g = 0; g < 4; ++g) {
        atomicAdd(&sac[0 + g * 2], s1[g]); atomicAdd(&sac[0 + g * 2 + 1], q1[g]);
        atomicAdd(&sac[8 + g * 2], s2[g]); atomicAdd(&sac[8 + g * 2 + 1], q2[g]);
        atomicAdd(&sac[16 + g * 2], s7[g]); atomicAdd(&sac[16 + g * 2 + 1], q7[g]);
    }
    __syncthreads();
    if (tid < 24) {
        const int gl = tid >> 3, rem = tid & 7;
        const int stage = (gl == 0) ? 0 : (gl == 1 ? 1 : 6);
        atomicAdd(&p.sraw[stage * 32 + b * 8 + rem], sac[tid]);
    }
}

// ---------------- finalize one GN stage ----------------
__global__ void k_fin(P p, int stage, float cnt) {
    const int t = threadIdx.x;
    if (t < 16) {
        const float s = p.sraw[stage * 32 + t * 2];
        const float q = p.sraw[stage * 32 + t * 2 + 1];
        const float m = s / cnt;
        const float var = q / cnt - m * m;
        p.sfin[stage * 32 + t * 2] = m;
        p.sfin[stage * 32 + t * 2 + 1] = rsqrtf(var + EPSF);
    }
}

// ---------------- attention pool core ----------------
// xrow[c] = f_concat[c][mrow] (this lane's neighbor column). Result -> aggb[0..15].
__device__ __forceinline__ void att_core(const float* Wa, const float* xrow,
                                         float* aggb, int lane) {
    const int mrow = lane & 15;
    const int gg   = lane >> 4;

    v16h a, bm;
#pragma unroll
    for (int j = 0; j < 8; ++j) {                     // A 16x32 f16, contiguous k = 16*gg + j, K padded 16->32
        a[j]     = gg ? (_Float16)0.f : (_Float16)Wa[mrow * 16 + j];
        a[j + 8] = gg ? (_Float16)0.f : (_Float16)Wa[mrow * 16 + j + 8];
    }
#pragma unroll
    for (int j = 0; j < 16; ++j)                      // B 32x16 f16 layout (rows 16..31 zero)
        bm[j] = gg ? (_Float16)0.f : (_Float16)xrow[j];

    v8f cz = {0.f,0.f,0.f,0.f,0.f,0.f,0.f,0.f};
    v8f sc = __builtin_amdgcn_wmma_f32_16x16x32_f16(false, a, false, bm,
                                                    (short)0, cz, false, false);

    float sv[8];
#pragma unroll
    for (int r = 0; r < 8; ++r) sv[r] = sc[r];

    // per row o = r + 8*gg: softmax over k (the 16 lanes of this row), then agg
#pragma unroll
    for (int r = 0; r < 8; ++r) {
        const float mx = rowmax16(sv[r]);
        const float e  = __expf(sv[r] - mx);
        const float s  = rowsum16(e);
        sv[r] = rowsum16((e / s) * xrow[r + 8 * gg]);
    }
#pragma unroll
    for (int r = 0; r < 8; ++r)
        if (mrow == r) aggb[r + 8 * gg] = sv[r];
}

// ---------------- stage 2: att_pool1 -> f_agg_pre, GN3 stats; GN4 stats ----------------
__global__ void k_stage2(P p) {
    __shared__ __align__(16) float xcatT[8][16][16];  // [wave][neighbor k][channel c]
    __shared__ __align__(16) float aggb[8][16];
    __shared__ float acc16[16];                       // [0..7]=GN3 s/q x4, [8..15]=GN4 s/q x4
    const int tid  = threadIdx.x;
    const int wv   = tid >> 5;
    const int lane = tid & 31;
    if (tid < 16) acc16[tid] = 0.f;
    __syncthreads();

    const int b = blockIdx.x >> 13;
    const int n = ((blockIdx.x & 8191) << 3) + wv;

    if (lane < 16) {                                  // gather f_pc (GN1 + lrelu) -> ch 0..7
        const int k = lane;
        const int idx = p.nidx[(b * NN + n) * KK + k];
        float fv[8];
#pragma unroll
        for (int c = 0; c < 8; ++c) {
            const int g = c >> 1;
            const float m = p.sfin[0 * 32 + b * 8 + g * 2];
            const float r = p.sfin[0 * 32 + b * 8 + g * 2 + 1];
            const float vp = p.fpc[(b * 8 + c) * NN + idx];
            fv[c] = lrelu((vp - m) * r * p.g_mlp1[c] + p.be_mlp1[c]);
        }
        *(float4*)&xcatT[wv][k][0] = make_float4(fv[0], fv[1], fv[2], fv[3]);
        *(float4*)&xcatT[wv][k][4] = make_float4(fv[4], fv[5], fv[6], fv[7]);
    } else {                                          // recompute f_xyz (GN2 + lrelu) -> ch 8..15
        const int k = lane - 16;
        const int idx = p.nidx[(b * NN + n) * KK + k];
        const float px = p.xyz[(b * 3 + 0) * NN + n];
        const float py = p.xyz[(b * 3 + 1) * NN + n];
        const float pz = p.xyz[(b * 3 + 2) * NN + n];
        const float nx = p.xyz[(b * 3 + 0) * NN + idx];
        const float ny = p.xyz[(b * 3 + 1) * NN + idx];
        const float nz = p.xyz[(b * 3 + 2) * NN + idx];
        const float rx = nx - px, ry = ny - py, rz = nz - pz;
        const float dist = sqrtf(rx * rx + ry * ry + rz * rz);
        const float frel[10] = {dist, rx, ry, rz, px, py, pz, nx, ny, nz};
        float fv[8];
#pragma unroll
        for (int c2 = 0; c2 < 8; ++c2) {
            float v = p.b_lfa1[c2];
#pragma unroll
            for (int j = 0; j < 10; ++j) v += p.W_lfa1[c2 * 10 + j] * frel[j];
            const int g = c2 >> 1;
            const float m = p.sfin[1 * 32 + b * 8 + g * 2];
            const float r = p.sfin[1 * 32 + b * 8 + g * 2 + 1];
            fv[c2] = lrelu((v - m) * r * p.g_lfa1[c2] + p.be_lfa1[c2]);
        }
        *(float4*)&xcatT[wv][k][8]  = make_float4(fv[0], fv[1], fv[2], fv[3]);
        *(float4*)&xcatT[wv][k][12] = make_float4(fv[4], fv[5], fv[6], fv[7]);
    }
    __syncthreads();

    const int mrow = lane & 15, gg = lane >> 4;
    const float4 R0 = *(const float4*)&xcatT[wv][mrow][0];
    const float4 R1 = *(const float4*)&xcatT[wv][mrow][4];
    const float4 R2 = *(const float4*)&xcatT[wv][mrow][8];
    const float4 R3 = *(const float4*)&xcatT[wv][mrow][12];
    const float xrow[16] = {R0.x,R0.y,R0.z,R0.w, R1.x,R1.y,R1.z,R1.w,
                            R2.x,R2.y,R2.z,R2.w, R3.x,R3.y,R3.z,R3.w};

    att_core(p.Wa1, xrow, aggb[wv], lane);

    // GN4 stats for f_xyz2_pre (values recomputed in stage 3); xrow[8..15] = f_xyz at k=mrow
    {
        float sl = 0.f, ql = 0.f, sh = 0.f, qh = 0.f;
#pragma unroll
        for (int i = 0; i < 4; ++i) {
            const int c2 = 4 * gg + i;
            float v = p.b_lfa2[c2];
#pragma unroll
            for (int c = 0; c < 8; ++c) v += p.W_lfa2[c2 * 8 + c] * xrow[8 + c];
            if (i < 2) { sl += v; ql += v * v; } else { sh += v; qh += v * v; }
        }
        sl = rowsum16(sl); ql = rowsum16(ql); sh = rowsum16(sh); qh = rowsum16(qh);
        if (mrow == 0) {                              // groups 2*gg, 2*gg+1
            atomicAdd(&acc16[8 + (2 * gg) * 2],     sl);
            atomicAdd(&acc16[8 + (2 * gg) * 2 + 1], ql);
            atomicAdd(&acc16[8 + (2 * gg + 1) * 2],     sh);
            atomicAdd(&acc16[8 + (2 * gg + 1) * 2 + 1], qh);
        }
    }
    __syncthreads();

    if (lane < 8) {                                   // f_agg_pre = Wm1 @ agg + bm1
        const int o = lane;
        const float4 A0 = *(const float4*)&aggb[wv][0];
        const float4 A1 = *(const float4*)&aggb[wv][4];
        const float4 A2 = *(const float4*)&aggb[wv][8];
        const float4 A3 = *(const float4*)&aggb[wv][12];
        const float ag[16] = {A0.x,A0.y,A0.z,A0.w, A1.x,A1.y,A1.z,A1.w,
                              A2.x,A2.y,A2.z,A2.w, A3.x,A3.y,A3.z,A3.w};
        float v = p.bm1[o];
#pragma unroll
        for (int c = 0; c < 16; ++c) v += p.Wm1[o * 16 + c] * ag[c];
        p.fagg[(b * 8 + o) * NN + n] = v;
        atomicAdd(&acc16[(o >> 1) * 2], v);
        atomicAdd(&acc16[(o >> 1) * 2 + 1], v * v);
    }
    __syncthreads();
    if (tid < 8)       atomicAdd(&p.sraw[2 * 32 + b * 8 + tid], acc16[tid]);
    else if (tid < 16) atomicAdd(&p.sraw[3 * 32 + b * 8 + (tid - 8)], acc16[tid]);
}

// ---------------- stage 3: att_pool2 -> f_agg2_pre, GN5 stats ----------------
__global__ void k_stage3(P p) {
    __shared__ __align__(16) float xcatT[8][16][16];
    __shared__ __align__(16) float aggb[8][16];
    __shared__ float acc8[8];
    const int tid  = threadIdx.x;
    const int wv   = tid >> 5;
    const int lane = tid & 31;
    if (tid < 8) acc8[tid] = 0.f;
    __syncthreads();

    const int b = blockIdx.x >> 13;
    const int n = ((blockIdx.x & 8191) << 3) + wv;

    if (lane < 16) {                                  // gather f_agg (GN3 + lrelu) -> ch 0..7
        const int k = lane;
        const int idx = p.nidx[(b * NN + n) * KK + k];
        float fv[8];
#pragma unroll
        for (int c = 0; c < 8; ++c) {
            const int g = c >> 1;
            const float m = p.sfin[2 * 32 + b * 8 + g * 2];
            const float r = p.sfin[2 * 32 + b * 8 + g * 2 + 1];
            const float vp = p.fagg[(b * 8 + c) * NN + idx];
            fv[c] = lrelu((vp - m) * r * p.gm1[c] + p.bem1[c]);
        }
        *(float4*)&xcatT[wv][k][0] = make_float4(fv[0], fv[1], fv[2], fv[3]);
        *(float4*)&xcatT[wv][k][4] = make_float4(fv[4], fv[5], fv[6], fv[7]);
    } else {                                          // f_xyz -> f_xyz2 (GN4 + lrelu) -> ch 8..15
        const int k = lane - 16;
        const int idx = p.nidx[(b * NN + n) * KK + k];
        const float px = p.xyz[(b * 3 + 0) * NN + n];
        const float py = p.xyz[(b * 3 + 1) * NN + n];
        const float pz = p.xyz[(b * 3 + 2) * NN + n];
        const float nx = p.xyz[(b * 3 + 0) * NN + idx];
        const float ny = p.xyz[(b * 3 + 1) * NN + idx];
        const float nz = p.xyz[(b * 3 + 2) * NN + idx];
        const float rx = nx - px, ry = ny - py, rz = nz - pz;
        const float dist = sqrtf(rx * rx + ry * ry + rz * rz);
        const float frel[10] = {dist, rx, ry, rz, px, py, pz, nx, ny, nz};
        float fx[8];
#pragma unroll
        for (int c2 = 0; c2 < 8; ++c2) {
            float v = p.b_lfa1[c2];
#pragma unroll
            for (int j = 0; j < 10; ++j) v += p.W_lfa1[c2 * 10 + j] * frel[j];
            const int g = c2 >> 1;
            const float m = p.sfin[1 * 32 + b * 8 + g * 2];
            const float r = p.sfin[1 * 32 + b * 8 + g * 2 + 1];
            fx[c2] = lrelu((v - m) * r * p.g_lfa1[c2] + p.be_lfa1[c2]);
        }
        float fv[8];
#pragma unroll
        for (int c2 = 0; c2 < 8; ++c2) {
            float v = p.b_lfa2[c2];
#pragma unroll
            for (int c = 0; c < 8; ++c) v += p.W_lfa2[c2 * 8 + c] * fx[c];
            const int g = c2 >> 1;
            const float m = p.sfin[3 * 32 + b * 8 + g * 2];
            const float r = p.sfin[3 * 32 + b * 8 + g * 2 + 1];
            fv[c2] = lrelu((v - m) * r * p.g_lfa2[c2] + p.be_lfa2[c2]);
        }
        *(float4*)&xcatT[wv][k][8]  = make_float4(fv[0], fv[1], fv[2], fv[3]);
        *(float4*)&xcatT[wv][k][12] = make_float4(fv[4], fv[5], fv[6], fv[7]);
    }
    __syncthreads();

    const int mrow = lane & 15;
    const float4 R0 = *(const float4*)&xcatT[wv][mrow][0];
    const float4 R1 = *(const float4*)&xcatT[wv][mrow][4];
    const float4 R2 = *(const float4*)&xcatT[wv][mrow][8];
    const float4 R3 = *(const float4*)&xcatT[wv][mrow][12];
    const float xrow[16] = {R0.x,R0.y,R0.z,R0.w, R1.x,R1.y,R1.z,R1.w,
                            R2.x,R2.y,R2.z,R2.w, R3.x,R3.y,R3.z,R3.w};

    att_core(p.Wa2, xrow, aggb[wv], lane);
    __syncthreads();

    if (lane < 16) {                                  // f_agg2_pre = Wm2 @ agg2 + bm2
        const int o = lane;
        const float4 A0 = *(const float4*)&aggb[wv][0];
        const float4 A1 = *(const float4*)&aggb[wv][4];
        const float4 A2 = *(const float4*)&aggb[wv][8];
        const float4 A3 = *(const float4*)&aggb[wv][12];
        const float ag[16] = {A0.x,A0.y,A0.z,A0.w, A1.x,A1.y,A1.z,A1.w,
                              A2.x,A2.y,A2.z,A2.w, A3.x,A3.y,A3.z,A3.w};
        float v = p.bm2[o];
#pragma unroll
        for (int c = 0; c < 16; ++c) v += p.Wm2[o * 16 + c] * ag[c];
        p.fagg2[(b * 16 + o) * NN + n] = v;
        atomicAdd(&acc8[(o >> 2) * 2], v);
        atomicAdd(&acc8[(o >> 2) * 2 + 1], v * v);
    }
    __syncthreads();
    if (tid < 8) atomicAdd(&p.sraw[4 * 32 + b * 8 + tid], acc8[tid]);
}

// ---------------- stage 4: f_out_pre = W_mlp2 @ GN5(f_agg2), GN6 stats ----------------
__global__ void k_stage4(P p) {
    __shared__ float acc8[8];
    const int tid = threadIdx.x;
    if (tid < 8) acc8[tid] = 0.f;
    __syncthreads();

    const int b = blockIdx.x >> 8;
    const int n = ((blockIdx.x & 255) << 8) + tid;

    float fa[16];
#pragma unroll
    for (int c = 0; c < 16; ++c) {
        const int g = c >> 2;
        const float m = p.sfin[4 * 32 + b * 8 + g * 2];
        const float r = p.sfin[4 * 32 + b * 8 + g * 2 + 1];
        const float vp = p.fagg2[(b * 16 + c) * NN + n];
        fa[c] = lrelu((vp - m) * r * p.gm2[c] + p.bem2[c]);
    }
    float s[4] = {0,0,0,0}, q[4] = {0,0,0,0};
#pragma unroll
    for (int o = 0; o < 32; ++o) {
        float v = p.b_mlp2[o];
#pragma unroll
        for (int c = 0; c < 16; ++c) v += p.W_mlp2[o * 16 + c] * fa[c];
        p.fout[(b * 32 + o) * NN + n] = v;
        s[o >> 3] += v; q[o >> 3] += v * v;
    }
#pragma unroll
    for (int g = 0; g < 4; ++g) {
        atomicAdd(&acc8[g * 2], s[g]); atomicAdd(&acc8[g * 2 + 1], q[g]);
    }
    __syncthreads();
    if (tid < 8) atomicAdd(&p.sraw[5 * 32 + b * 8 + tid], acc8[tid]);
}

// ---------------- stage 5: out = lrelu(GN6(f_out) + GN7(shortcut)), float4 ----------------
__global__ void k_stage5(P p) {
    const int t = blockIdx.x * 256 + threadIdx.x;
    const int i = t * 4;
    const int o = (i >> 16) & 31;
    const int b = i >> 21;
    const int g = o >> 3;
    const float m6 = p.sfin[5 * 32 + b * 8 + g * 2];
    const float r6 = p.sfin[5 * 32 + b * 8 + g * 2 + 1];
    const float m7 = p.sfin[6 * 32 + b * 8 + g * 2];
    const float r7 = p.sfin[6 * 32 + b * 8 + g * 2 + 1];
    const float ga = p.g_mlp2[o], ba = p.be_mlp2[o];
    const float gs = p.g_skip[o], bs = p.be_skip[o];
    const float4 fo = *(const float4*)&p.fout[i];
    const float4 sh = *(const float4*)&p.shc[i];
    float4 r;
    r.x = lrelu((fo.x - m6) * r6 * ga + ba + (sh.x - m7) * r7 * gs + bs);
    r.y = lrelu((fo.y - m6) * r6 * ga + ba + (sh.y - m7) * r7 * gs + bs);
    r.z = lrelu((fo.z - m6) * r6 * ga + ba + (sh.z - m7) * r7 * gs + bs);
    r.w = lrelu((fo.w - m6) * r6 * ga + ba + (sh.w - m7) * r7 * gs + bs);
    *(float4*)&p.out[i] = r;
}

extern "C" void kernel_launch(void* const* d_in, const int* in_sizes, int n_in,
                              void* d_out, int out_size, void* d_ws, size_t ws_size,
                              hipStream_t stream) {
    (void)in_sizes; (void)n_in; (void)out_size; (void)ws_size;
    P p;
    p.feature = (const float*)d_in[0];
    p.xyz     = (const float*)d_in[1];
    p.nidx    = (const int*)  d_in[2];
    p.W_mlp1 = (const float*)d_in[3];  p.b_mlp1 = (const float*)d_in[4];
    p.g_mlp1 = (const float*)d_in[5];  p.be_mlp1 = (const float*)d_in[6];
    p.W_lfa1 = (const float*)d_in[7];  p.b_lfa1 = (const float*)d_in[8];
    p.g_lfa1 = (const float*)d_in[9];  p.be_lfa1 = (const float*)d_in[10];
    p.Wa1 = (const float*)d_in[11];    p.Wm1 = (const float*)d_in[12];
    p.bm1 = (const float*)d_in[13];    p.gm1 = (const float*)d_in[14];
    p.bem1 = (const float*)d_in[15];
    p.W_lfa2 = (const float*)d_in[16]; p.b_lfa2 = (const float*)d_in[17];
    p.g_lfa2 = (const float*)d_in[18]; p.be_lfa2 = (const float*)d_in[19];
    p.Wa2 = (const float*)d_in[20];    p.Wm2 = (const float*)d_in[21];
    p.bm2 = (const float*)d_in[22];    p.gm2 = (const float*)d_in[23];
    p.bem2 = (const float*)d_in[24];
    p.W_mlp2 = (const float*)d_in[25]; p.b_mlp2 = (const float*)d_in[26];
    p.g_mlp2 = (const float*)d_in[27]; p.be_mlp2 = (const float*)d_in[28];
    p.W_skip = (const float*)d_in[29]; p.b_skip = (const float*)d_in[30];
    p.g_skip = (const float*)d_in[31]; p.be_skip = (const float*)d_in[32];

    float* ws = (float*)d_ws;
    p.sraw  = ws;                                // 224 floats
    p.sfin  = ws + 256;                          // 224 floats
    p.fpc   = ws + 512;                          // 4*8*65536
    p.shc   = p.fpc   + 4 * 8 * NN;              // 4*32*65536
    p.fagg  = p.shc   + 4 * 32 * NN;             // 4*8*65536
    p.fagg2 = p.fagg  + 4 * 8 * NN;              // 4*16*65536
    p.fout  = p.fagg2 + 4 * 16 * NN;             // 4*32*65536
    p.out   = (float*)d_out;

    hipMemsetAsync(p.sraw, 0, 224 * sizeof(float), stream);

    k_stage1<<<dim3(BB * 256), dim3(256), 0, stream>>>(p);
    k_fin<<<dim3(1), dim3(16), 0, stream>>>(p, 0, 2.f * NN);            // GN1
    k_fin<<<dim3(1), dim3(16), 0, stream>>>(p, 1, 2.f * NN * KK);       // GN2
    k_fin<<<dim3(1), dim3(16), 0, stream>>>(p, 6, 8.f * NN);            // GN7

    k_stage2<<<dim3(BB * 8192), dim3(256), 0, stream>>>(p);
    k_fin<<<dim3(1), dim3(16), 0, stream>>>(p, 2, 2.f * NN);            // GN3
    k_fin<<<dim3(1), dim3(16), 0, stream>>>(p, 3, 2.f * NN * KK);       // GN4

    k_stage3<<<dim3(BB * 8192), dim3(256), 0, stream>>>(p);
    k_fin<<<dim3(1), dim3(16), 0, stream>>>(p, 4, 4.f * NN);            // GN5

    k_stage4<<<dim3(BB * 256), dim3(256), 0, stream>>>(p);
    k_fin<<<dim3(1), dim3(16), 0, stream>>>(p, 5, 8.f * NN);            // GN6

    k_stage5<<<dim3(BB * 32 * NN / 256 / 4), dim3(256), 0, stream>>>(p);
}
